// PSIBlock_EMA_47931835024036
// MI455X (gfx1250) — matrix operand order repaired and checked by reading, not compiled, not run on hardware
//
#include <hip/hip_runtime.h>
#include <hip/hip_bf16.h>
#include <math.h>

typedef _Float16 v16h __attribute__((ext_vector_type(16)));
typedef _Float16 v8h  __attribute__((ext_vector_type(8)));
typedef float    v8f  __attribute__((ext_vector_type(8)));
typedef unsigned int u32x4 __attribute__((ext_vector_type(4)));
typedef int          i32x8 __attribute__((ext_vector_type(8)));
typedef int          i32x4 __attribute__((ext_vector_type(4)));

#define DECAY 0.95f
#define EPS   1e-5f

// LDS A-tile: 64 rows x 32 halves, TDM-padded +4 dwords per 16-dword row
// -> row stride 40 halves (80B) to avoid LDS bank conflicts.
#define A_ROW_STRIDE 40
#define A_TILE_HALVES (64 * A_ROW_STRIDE)

// ---------------------------------------------------------------------------
// TDM: DMA one 64x32 f16 tile of row-major A[M x K] into LDS (async, TENSORcnt)
// Descriptor layout per CDNA5 ISA (D# groups 0..3).
// This toolchain exposes the 6-arg builtin: (g0, g1, g2, g3, extra, cpol).
// ---------------------------------------------------------------------------
__device__ __forceinline__ void tdm_load_a_tile(const _Float16* gsrc,
                                                unsigned lds_byte_addr,
                                                int K, int Mtot) {
    unsigned long long ga = (unsigned long long)(size_t)gsrc;
    u32x4 g0;
    g0[0] = 1u;                                   // count=1 (valid user D#)
    g0[1] = lds_byte_addr;                        // lds_addr
    g0[2] = (unsigned)ga;                         // global_addr[31:0]
    g0[3] = (unsigned)((ga >> 32) & 0x01FFFFFFu)  // global_addr[56:32]
          | 0x80000000u;                          // type=2 ("image")
    i32x8 g1;
    g1[0] = (1 << 16)                             // data_size = 2 bytes
          | (1 << 20)                             // pad_enable
          | (3 << 22)                             // pad_interval: 16 dwords
          | (3 << 25);                            // pad_amount: 4 dwords
    g1[1] = (int)((K & 0xFFFF) << 16);            // tensor_dim0 lo16 (bits 63:48)
    g1[2] = (int)(((unsigned)K >> 16) | (((unsigned)Mtot & 0xFFFFu) << 16));
    g1[3] = (int)(((unsigned)Mtot >> 16) | (32u << 16));  // tile_dim0 = 32
    g1[4] = 64;                                   // tile_dim1 = 64 rows
    g1[5] = K;                                    // tensor_dim0_stride
    g1[6] = 0;
    g1[7] = 0;
    i32x4 z4; z4[0] = 0; z4[1] = 0; z4[2] = 0; z4[3] = 0;
    i32x8 z8; z8[0] = 0; z8[1] = 0; z8[2] = 0; z8[3] = 0;
              z8[4] = 0; z8[5] = 0; z8[6] = 0; z8[7] = 0;
    __builtin_amdgcn_tensor_load_to_lds(g0, g1, z4, z4, z8, 0);
}

// ---------------------------------------------------------------------------
// Fragment loaders (contiguous k = 16*(lane/16) + elem for A and B, wave32)
// ---------------------------------------------------------------------------

// A 16x32 f16 fragment from the padded LDS tile.
// lanes 0-15: row=lane,  elems 0..15 -> K=0..15
// lanes16-31: row=lane-16, elems 0..15 -> K=16..31
__device__ __forceinline__ v16h load_a_frag_lds(const _Float16* buf,
                                                int row0, int lane) {
    const int r  = row0 + (lane & 15);
    const int kb = (lane & 16) ? 16 : 0;
    const _Float16* p = buf + r * A_ROW_STRIDE + kb;
    v8h lo = *reinterpret_cast<const v8h*>(p);
    v8h hi = *reinterpret_cast<const v8h*>(p + 8);
    return __builtin_shufflevector(lo, hi, 0,1,2,3,4,5,6,7,8,9,10,11,12,13,14,15);
}

// B 32x16 f16 pre-packed: tile(nt,kt) = 32 lanes x 16 halves contiguous.
__device__ __forceinline__ v16h load_b_frag(const _Float16* __restrict__ Bp,
                                            int ktiles, int nt, int kt, int lane) {
    const size_t off = ((((size_t)nt * ktiles) + kt) * 32 + lane) * 16;
    return *reinterpret_cast<const v16h*>(Bp + off);
}

__device__ __forceinline__ v8f wmma_f16(v16h a, v16h b, v8f c) {
    return __builtin_amdgcn_wmma_f32_16x16x32_f16(false, a, false, b,
                                                  (short)0, c, false, false);
}

// ---------------------------------------------------------------------------
// Weight pack: fp32 row-major W[K x N] -> f16 B-fragment layout
// lanes 0-15: col n=lane, k = e ; lanes 16-31: col n=lane-16, k = 16 + e
// ---------------------------------------------------------------------------
__global__ __launch_bounds__(256) void k_pack_b(const float* __restrict__ W,
                                                _Float16* __restrict__ out,
                                                int K, int N) {
    const int idx    = blockIdx.x * blockDim.x + threadIdx.x;
    const int ktiles = K >> 5;
    const int total  = ktiles * (N >> 4) * 32;
    if (idx >= total) return;
    const int lane = idx & 31;
    const int tile = idx >> 5;          // tile = nt*ktiles + kt
    const int kt   = tile % ktiles;
    const int nt   = tile / ktiles;
    const int n    = (nt << 4) + (lane & 15);
    const int kb   = (kt << 5) + ((lane & 16) ? 16 : 0);
    v16h v;
    #pragma unroll
    for (int e = 0; e < 16; ++e)
        v[e] = (_Float16)W[(size_t)(kb + e) * N + n];
    *reinterpret_cast<v16h*>(out + (size_t)idx * 16) = v;
}

// ---------------------------------------------------------------------------
// LayerNorm (one block per row) -> f16 activations
// ---------------------------------------------------------------------------
__global__ __launch_bounds__(256) void k_layernorm(const float* __restrict__ x,
                                                   const float* __restrict__ w,
                                                   const float* __restrict__ b,
                                                   _Float16* __restrict__ out,
                                                   int D) {
    __shared__ float s1[256];
    __shared__ float s2[256];
    const int row = blockIdx.x;
    const int tid = threadIdx.x;
    const float* xr = x + (size_t)row * D;
    float s = 0.f, q = 0.f;
    for (int i = tid; i < D; i += 256) { float v = xr[i]; s += v; q += v * v; }
    s1[tid] = s; s2[tid] = q; __syncthreads();
    for (int off = 128; off > 0; off >>= 1) {
        if (tid < off) { s1[tid] += s1[tid + off]; s2[tid] += s2[tid + off]; }
        __syncthreads();
    }
    const float mu  = s1[0] / (float)D;
    const float var = s2[0] / (float)D - mu * mu;
    const float r   = rsqrtf(var + EPS);
    _Float16* orow = out + (size_t)row * D;
    for (int i = tid; i < D; i += 256)
        orow[i] = (_Float16)((xr[i] - mu) * r * w[i] + b[i]);
}

// ---------------------------------------------------------------------------
// Fused gate/value GEMM: gv = sigmoid(h@Wg+bg) * (h@Wv+bv)   (fp32 out)
// block = 8 waves -> 64(M) x 128(N); wave = 32x32 (2x2 WMMA tiles).
// A tile (64x32) DMA'd into double-buffered LDS by the TDM.
// ---------------------------------------------------------------------------
__global__ __launch_bounds__(256) void k_gate_value(const _Float16* __restrict__ h,
                                                    const _Float16* __restrict__ wg,
                                                    const _Float16* __restrict__ wv,
                                                    const float* __restrict__ bg,
                                                    const float* __restrict__ bv,
                                                    float* __restrict__ gv,
                                                    int M, int K, int N) {
    __shared__ __align__(16) _Float16 abuf[2][A_TILE_HALVES];
    const int lane = threadIdx.x & 31;
    const int wave = threadIdx.x >> 5;
    const int mblk = blockIdx.y * 64;
    const int mloc = (wave >> 2) * 32;
    const int n0   = blockIdx.x * 128 + (wave & 3) * 32;
    const int ktiles = K >> 5;
    const int nt0 = n0 >> 4;
    const unsigned lds[2] = { (unsigned)(size_t)(void*)&abuf[0][0],
                              (unsigned)(size_t)(void*)&abuf[1][0] };
    const _Float16* abase = h + (size_t)mblk * K;

    if (wave == 0) {
        tdm_load_a_tile(abase, lds[0], K, M);
        __builtin_amdgcn_s_wait_tensorcnt(0);
    }
    __syncthreads();

    v8f cg[2][2] = {};
    v8f cv[2][2] = {};
    for (int kt = 0; kt < ktiles; ++kt) {
        if (wave == 0 && kt + 1 < ktiles)
            tdm_load_a_tile(abase + ((kt + 1) << 5), lds[(kt + 1) & 1], K, M);

        const _Float16* at = &abuf[kt & 1][0];
        v16h a0 = load_a_frag_lds(at, mloc,      lane);
        v16h a1 = load_a_frag_lds(at, mloc + 16, lane);
        v16h g0 = load_b_frag(wg, ktiles, nt0,     kt, lane);
        v16h g1 = load_b_frag(wg, ktiles, nt0 + 1, kt, lane);
        v16h v0 = load_b_frag(wv, ktiles, nt0,     kt, lane);
        v16h v1 = load_b_frag(wv, ktiles, nt0 + 1, kt, lane);
        cg[0][0] = wmma_f16(a0, g0, cg[0][0]);
        cg[0][1] = wmma_f16(a0, g1, cg[0][1]);
        cg[1][0] = wmma_f16(a1, g0, cg[1][0]);
        cg[1][1] = wmma_f16(a1, g1, cg[1][1]);
        cv[0][0] = wmma_f16(a0, v0, cv[0][0]);
        cv[0][1] = wmma_f16(a0, v1, cv[0][1]);
        cv[1][0] = wmma_f16(a1, v0, cv[1][0]);
        cv[1][1] = wmma_f16(a1, v1, cv[1][1]);

        if (wave == 0 && kt + 1 < ktiles)
            __builtin_amdgcn_s_wait_tensorcnt(0);
        __syncthreads();
    }
    // C layout: elem r, lanes 0-15: M=r / lanes 16-31: M=8+r ; N = lane&15
    #pragma unroll
    for (int i = 0; i < 2; ++i)
        #pragma unroll
        for (int j = 0; j < 2; ++j)
            #pragma unroll
            for (int r = 0; r < 8; ++r) {
                const int m = mblk + mloc + 16 * i + r + ((lane & 16) ? 8 : 0);
                const int n = n0 + 16 * j + (lane & 15);
                const float g = cg[i][j][r] + bg[n];
                const float v = cv[i][j][r] + bv[n];
                gv[(size_t)m * N + n] = v * (1.0f / (1.0f + __expf(-g)));
            }
}

// ---------------------------------------------------------------------------
// EMA scan over S + residual: xn = x + mem   (one thread per (b,d) channel)
// ---------------------------------------------------------------------------
__global__ __launch_bounds__(256) void k_ema(const float* __restrict__ gv,
                                             const float* __restrict__ x,
                                             float* __restrict__ xn,
                                             int B, int S, int D) {
    const int idx = blockIdx.x * blockDim.x + threadIdx.x;
    if (idx >= B * D) return;
    const int b = idx / D;
    const int d = idx - b * D;
    const size_t base = (size_t)b * S * D + d;
    const float* g  = gv + base;
    const float* xp = x  + base;
    float*       xo = xn + base;
    float m = 0.f;
    for (int t = 0; t < S; ++t) {
        const size_t o = (size_t)t * D;
        m = DECAY * m + (1.0f - DECAY) * g[o];
        xo[o] = xp[o] + m;
    }
}

// ---------------------------------------------------------------------------
// FFN1 GEMM (TDM + WMMA) with exact-erf GELU epilogue -> f16
// ---------------------------------------------------------------------------
__global__ __launch_bounds__(256) void k_ffn1(const _Float16* __restrict__ A,
                                              const _Float16* __restrict__ Bw,
                                              const float* __restrict__ bias,
                                              _Float16* __restrict__ out,
                                              int M, int K, int N) {
    __shared__ __align__(16) _Float16 abuf[2][A_TILE_HALVES];
    const int lane = threadIdx.x & 31;
    const int wave = threadIdx.x >> 5;
    const int mblk = blockIdx.y * 64;
    const int mloc = (wave >> 2) * 32;
    const int n0   = blockIdx.x * 128 + (wave & 3) * 32;
    const int ktiles = K >> 5;
    const int nt0 = n0 >> 4;
    const unsigned lds[2] = { (unsigned)(size_t)(void*)&abuf[0][0],
                              (unsigned)(size_t)(void*)&abuf[1][0] };
    const _Float16* abase = A + (size_t)mblk * K;

    if (wave == 0) {
        tdm_load_a_tile(abase, lds[0], K, M);
        __builtin_amdgcn_s_wait_tensorcnt(0);
    }
    __syncthreads();

    v8f c[2][2] = {};
    for (int kt = 0; kt < ktiles; ++kt) {
        if (wave == 0 && kt + 1 < ktiles)
            tdm_load_a_tile(abase + ((kt + 1) << 5), lds[(kt + 1) & 1], K, M);

        const _Float16* at = &abuf[kt & 1][0];
        v16h a0 = load_a_frag_lds(at, mloc,      lane);
        v16h a1 = load_a_frag_lds(at, mloc + 16, lane);
        v16h b0 = load_b_frag(Bw, ktiles, nt0,     kt, lane);
        v16h b1 = load_b_frag(Bw, ktiles, nt0 + 1, kt, lane);
        c[0][0] = wmma_f16(a0, b0, c[0][0]);
        c[0][1] = wmma_f16(a0, b1, c[0][1]);
        c[1][0] = wmma_f16(a1, b0, c[1][0]);
        c[1][1] = wmma_f16(a1, b1, c[1][1]);

        if (wave == 0 && kt + 1 < ktiles)
            __builtin_amdgcn_s_wait_tensorcnt(0);
        __syncthreads();
    }
    #pragma unroll
    for (int i = 0; i < 2; ++i)
        #pragma unroll
        for (int j = 0; j < 2; ++j)
            #pragma unroll
            for (int r = 0; r < 8; ++r) {
                const int m = mblk + mloc + 16 * i + r + ((lane & 16) ? 8 : 0);
                const int n = n0 + 16 * j + (lane & 15);
                const float t  = c[i][j][r] + bias[n];
                const float ge = 0.5f * t * (1.0f + erff(t * 0.70710678118654752f));
                out[(size_t)m * N + n] = (_Float16)ge;
            }
}

// ---------------------------------------------------------------------------
// FFN2 GEMM (TDM + WMMA) with residual add -> fp32 output
// ---------------------------------------------------------------------------
__global__ __launch_bounds__(256) void k_ffn2(const _Float16* __restrict__ A,
                                              const _Float16* __restrict__ Bw,
                                              const float* __restrict__ bias,
                                              const float* __restrict__ res,
                                              float* __restrict__ out,
                                              int M, int K, int N) {
    __shared__ __align__(16) _Float16 abuf[2][A_TILE_HALVES];
    const int lane = threadIdx.x & 31;
    const int wave = threadIdx.x >> 5;
    const int mblk = blockIdx.y * 64;
    const int mloc = (wave >> 2) * 32;
    const int n0   = blockIdx.x * 128 + (wave & 3) * 32;
    const int ktiles = K >> 5;
    const int nt0 = n0 >> 4;
    const unsigned lds[2] = { (unsigned)(size_t)(void*)&abuf[0][0],
                              (unsigned)(size_t)(void*)&abuf[1][0] };
    const _Float16* abase = A + (size_t)mblk * K;

    if (wave == 0) {
        tdm_load_a_tile(abase, lds[0], K, M);
        __builtin_amdgcn_s_wait_tensorcnt(0);
    }
    __syncthreads();

    v8f c[2][2] = {};
    for (int kt = 0; kt < ktiles; ++kt) {
        if (wave == 0 && kt + 1 < ktiles)
            tdm_load_a_tile(abase + ((kt + 1) << 5), lds[(kt + 1) & 1], K, M);

        const _Float16* at = &abuf[kt & 1][0];
        v16h a0 = load_a_frag_lds(at, mloc,      lane);
        v16h a1 = load_a_frag_lds(at, mloc + 16, lane);
        v16h b0 = load_b_frag(Bw, ktiles, nt0,     kt, lane);
        v16h b1 = load_b_frag(Bw, ktiles, nt0 + 1, kt, lane);
        c[0][0] = wmma_f16(a0, b0, c[0][0]);
        c[0][1] = wmma_f16(a0, b1, c[0][1]);
        c[1][0] = wmma_f16(a1, b0, c[1][0]);
        c[1][1] = wmma_f16(a1, b1, c[1][1]);

        if (wave == 0 && kt + 1 < ktiles)
            __builtin_amdgcn_s_wait_tensorcnt(0);
        __syncthreads();
    }
    #pragma unroll
    for (int i = 0; i < 2; ++i)
        #pragma unroll
        for (int j = 0; j < 2; ++j)
            #pragma unroll
            for (int r = 0; r < 8; ++r) {
                const int m = mblk + mloc + 16 * i + r + ((lane & 16) ? 8 : 0);
                const int n = n0 + 16 * j + (lane & 15);
                const size_t o = (size_t)m * N + n;
                out[o] = res[o] + c[i][j][r] + bias[n];
            }
}

// ---------------------------------------------------------------------------
// Host-side orchestration
// ---------------------------------------------------------------------------
extern "C" void kernel_launch(void* const* d_in, const int* in_sizes, int n_in,
                              void* d_out, int out_size, void* d_ws, size_t ws_size,
                              hipStream_t stream) {
    (void)in_sizes; (void)n_in; (void)out_size; (void)ws_size;
    const int B = 4, S = 4096, D = 1024, H = 4096;
    const int M = B * S;                     // 16384 rows

    const float* x       = (const float*)d_in[0];
    const float* n1w     = (const float*)d_in[1];
    const float* n1b     = (const float*)d_in[2];
    const float* n2w     = (const float*)d_in[3];
    const float* n2b     = (const float*)d_in[4];
    const float* gate_w  = (const float*)d_in[5];
    const float* gate_b  = (const float*)d_in[6];
    const float* value_w = (const float*)d_in[7];
    const float* value_b = (const float*)d_in[8];
    const float* ffn_w1  = (const float*)d_in[9];
    const float* ffn_b1  = (const float*)d_in[10];
    const float* ffn_w2  = (const float*)d_in[11];
    const float* ffn_b2  = (const float*)d_in[12];
    float* out = (float*)d_out;

    char* ws = (char*)d_ws;
    const size_t MB = 1024ull * 1024ull;
    _Float16* h16 = (_Float16*)(ws + 0);          //  32 MB  [M,D] f16
    float*    gv  = (float*)   (ws + 32  * MB);   //  64 MB  [M,D] f32
    float*    xn  = (float*)   (ws + 96  * MB);   //  64 MB  [M,D] f32
    _Float16* h2  = (_Float16*)(ws + 160 * MB);   //  32 MB  [M,D] f16
    _Float16* a1  = (_Float16*)(ws + 192 * MB);   // 128 MB  [M,H] f16
    _Float16* wgp = (_Float16*)(ws + 320 * MB);   //   2 MB packed
    _Float16* wvp = (_Float16*)(ws + 322 * MB);   //   2 MB packed
    _Float16* w1p = (_Float16*)(ws + 324 * MB);   //   8 MB packed
    _Float16* w2p = (_Float16*)(ws + 332 * MB);   //   8 MB packed

    // 1) pack weights fp32 -> f16 WMMA B-fragment layout
    {
        int lanesDD = (D >> 5) * (D >> 4) * 32;   // 65536
        int lanesDH = (D >> 5) * (H >> 4) * 32;   // 262144
        int lanesHD = (H >> 5) * (D >> 4) * 32;   // 262144
        k_pack_b<<<(lanesDD + 255) / 256, 256, 0, stream>>>(gate_w,  wgp, D, D);
        k_pack_b<<<(lanesDD + 255) / 256, 256, 0, stream>>>(value_w, wvp, D, D);
        k_pack_b<<<(lanesDH + 255) / 256, 256, 0, stream>>>(ffn_w1,  w1p, D, H);
        k_pack_b<<<(lanesHD + 255) / 256, 256, 0, stream>>>(ffn_w2,  w2p, H, D);
    }

    // 2) LN1 -> h (f16)
    k_layernorm<<<M, 256, 0, stream>>>(x, n1w, n1b, h16, D);

    // 3) gv = sigmoid(h@Wg+bg) * (h@Wv+bv)
    k_gate_value<<<dim3(D / 128, M / 64), 256, 0, stream>>>(h16, wgp, wvp,
                                                            gate_b, value_b, gv,
                                                            M, D, D);

    // 4) EMA scan + residual: xn = x + mem
    k_ema<<<(B * D + 255) / 256, 256, 0, stream>>>(gv, x, xn, B, S, D);

    // 5) LN2 -> h2 (f16)
    k_layernorm<<<M, 256, 0, stream>>>(xn, n2w, n2b, h2, D);

    // 6) a1 = gelu(h2@W1 + b1)  (f16)
    k_ffn1<<<dim3(H / 128, M / 64), 256, 0, stream>>>(h2, w1p, ffn_b1, a1, M, D, H);

    // 7) out = xn + a1@W2 + b2
    k_ffn2<<<dim3(D / 128, M / 64), 256, 0, stream>>>(a1, w2p, ffn_b2, xn, out, M, H, D);
}
